// nearAttention_80161269613015
// MI455X (gfx1250) — compile-verified, matrix operand order checked by reading, not run on hardware
//
#include <hip/hip_runtime.h>
#include <hip/hip_bf16.h>
#include <math.h>

#define B_ 4
#define L_ 1024
#define D_ 512
#define H_ 8
#define DH_ 64

typedef __bf16 bf16_t;
typedef __attribute__((ext_vector_type(16))) __bf16 v16bf;
typedef __attribute__((ext_vector_type(8)))  __bf16 v8bf;
typedef __attribute__((ext_vector_type(8)))  float  v8f;

// ---- workspace layout (bytes) ----
static const size_t NROW   = (size_t)B_ * H_ * L_;           // 32768 head-rows
static const size_t NBHLD  = NROW * DH_;                      // 2,097,152 elems
static const size_t WS_KB16  = 0;                             // bf16 K, head-major
static const size_t WS_QB16  = WS_KB16 + NBHLD * 2;           // bf16 Q, head-major
static const size_t WS_SQK   = WS_QB16 + NBHLD * 2;           // f32 ||k||^2 per row
static const size_t WS_SQQ   = WS_SQK + NROW * 4;             // f32 ||q||^2 per row
static const size_t WS_STATS = WS_SQQ + NROW * 4;             // 32 f32: [vS1,vS2]x8 then [sS1,sS2]x8
static const size_t WS_NEED  = WS_STATS + 32 * 4;

// d_out layout (elements): w [B,H,L,L] f32 | bx [B,L] i32 | Vn [B,H,L,d] f32
static const size_t W_ELEMS  = (size_t)B_ * H_ * L_ * L_;     // 33,554,432
static const size_t BX_ELEMS = (size_t)B_ * L_;               // 4096

// ------------------------------------------------------------------
__global__ __launch_bounds__(32) void init_stats_k(float* stats) {
    if (threadIdx.x < 32) stats[threadIdx.x] = 0.0f;
}

// One 64-thread block per (sel, b, h, l): convert f32 -> bf16 head-major, reduce sq-norm.
__global__ __launch_bounds__(64) void pack_bf16_k(const float* __restrict__ Q,
                                                  const float* __restrict__ K,
                                                  bf16_t* __restrict__ Qb,
                                                  bf16_t* __restrict__ Kb,
                                                  float* __restrict__ sqq,
                                                  float* __restrict__ sqk) {
    __shared__ float red[64];
    int bid = blockIdx.x;
    const int NR = (int)NROW;
    int sel = (bid >= NR) ? 1 : 0;             // 0 = Q, 1 = K
    int row = sel ? bid - NR : bid;            // (b*H + h)*L + l
    int b = row / (H_ * L_);
    int h = (row / L_) % H_;
    int l = row % L_;
    int k = threadIdx.x;

    const float* src = sel ? K : Q;
    float v = src[((size_t)(b * L_ + l)) * D_ + h * DH_ + k];
    (sel ? Kb : Qb)[(size_t)row * DH_ + k] = (bf16_t)v;

    red[k] = v * v;
    __syncthreads();
    for (int s = 32; s > 0; s >>= 1) {
        if (k < s) red[k] += red[k + s];
        __syncthreads();
    }
    if (k == 0) (sel ? sqk : sqq)[row] = red[0];
}

// V batch stats per head (reduce over B, L, d) via LDS atomics -> global atomics.
__global__ __launch_bounds__(256) void v_stats_k(const float* __restrict__ V,
                                                 float* __restrict__ stats) {
    __shared__ float ls[16];
    if (threadIdx.x < 16) ls[threadIdx.x] = 0.0f;
    __syncthreads();
    const size_t N = (size_t)B_ * L_ * D_;
    for (size_t idx = (size_t)blockIdx.x * blockDim.x + threadIdx.x; idx < N;
         idx += (size_t)gridDim.x * blockDim.x) {
        int dd = (int)(idx % D_);
        int h = dd >> 6;
        float v = V[idx];
        atomicAdd(&ls[2 * h], v);
        atomicAdd(&ls[2 * h + 1], v * v);
    }
    __syncthreads();
    if (threadIdx.x < 16) atomicAdd(&stats[threadIdx.x], ls[threadIdx.x]);
}

// Vn = (V - mean) * rsqrt(var + 64)   (eps = d, preserved source bug), head-major out.
__global__ __launch_bounds__(256) void v_norm_k(const float* __restrict__ V,
                                                const float* __restrict__ stats,
                                                float* __restrict__ vn_out) {
    size_t idx = (size_t)blockIdx.x * blockDim.x + threadIdx.x;
    if (idx >= (size_t)B_ * L_ * D_) return;
    int b = (int)(idx / ((size_t)L_ * D_));
    int r = (int)(idx % ((size_t)L_ * D_));
    int l = r / D_;
    int dd = r % D_;
    int h = dd >> 6;
    int k = dd & 63;
    const float Nh = (float)(B_ * L_ * DH_);
    float mean = stats[2 * h] / Nh;
    float var = stats[2 * h + 1] / Nh - mean * mean;
    float inv = rsqrtf(var + 64.0f);
    vn_out[(((size_t)(b * H_ + h)) * L_ + l) * DH_ + k] = (V[idx] - mean) * inv;
}

__global__ __launch_bounds__(256) void bx_copy_k(const int* __restrict__ bx,
                                                 int* __restrict__ out) {
    int t = blockIdx.x * blockDim.x + threadIdx.x;
    if (t < (int)BX_ELEMS) out[t] = bx[t];
}

// ------------------------------------------------------------------
// Batched WMMA GEMM: per (b,h) cross[i][j] = sum_k Kh[i,k]*Qh[j,k], then
// score = -sqrt(max(||k_i||^2 + ||q_j||^2 - 2*cross, 0)), fused per-head stats.
// Block = 256 thr (8 waves), tile = 128x128 output; wave -> 16 rows x 128 cols.
__global__ __launch_bounds__(256) void gemm_scores_k(const bf16_t* __restrict__ Kb,
                                                     const bf16_t* __restrict__ Qb,
                                                     const float* __restrict__ sqk,
                                                     const float* __restrict__ sqq,
                                                     float* __restrict__ scores,
                                                     float* __restrict__ sstats) {
    __shared__ float red[256];
    const int bh   = blockIdx.z;                 // b*H + h
    const int it   = blockIdx.y;
    const int jt   = blockIdx.x;
    const int wave = threadIdx.x >> 5;
    const int lane = threadIdx.x & 31;
    const int m    = lane & 15;
    const int kbse = (lane < 16) ? 0 : 8;        // 16-bit A/B fragment K split
    const int i0   = it * 128 + wave * 16;
    const int j0   = jt * 128;

    const bf16_t* Krow  = Kb + ((size_t)(bh * L_ + i0 + m)) * DH_;
    const bf16_t* Qbase = Qb + ((size_t)bh * L_) * DH_;

    v8f acc[8];
    const v8f vzero = {0.f, 0.f, 0.f, 0.f, 0.f, 0.f, 0.f, 0.f};
#pragma unroll
    for (int t = 0; t < 8; ++t) acc[t] = vzero;

#pragma unroll
    for (int kt = 0; kt < DH_; kt += 32) {
        v8bf alo = *(const v8bf*)(Krow + kt + kbse);
        v8bf ahi = *(const v8bf*)(Krow + kt + kbse + 16);
        v16bf A = __builtin_shufflevector(alo, ahi, 0, 1, 2, 3, 4, 5, 6, 7,
                                          8, 9, 10, 11, 12, 13, 14, 15);
#pragma unroll
        for (int t = 0; t < 8; ++t) {
            const bf16_t* Qrow = Qbase + (size_t)(j0 + t * 16 + m) * DH_;
            v8bf blo = *(const v8bf*)(Qrow + kt + kbse);
            v8bf bhi = *(const v8bf*)(Qrow + kt + kbse + 16);
            v16bf Bf = __builtin_shufflevector(blo, bhi, 0, 1, 2, 3, 4, 5, 6, 7,
                                               8, 9, 10, 11, 12, 13, 14, 15);
            acc[t] = __builtin_amdgcn_wmma_f32_16x16x32_bf16(
                false, A, false, Bf, (short)0, acc[t], false, false);
        }
    }

    // Epilogue: C/D layout -> n = lane&15, M = r + (lane>=16 ? 8 : 0)
    const int n    = lane & 15;
    const int mofs = (lane < 16) ? 0 : 8;
    float lsum = 0.0f, lsum2 = 0.0f;
#pragma unroll
    for (int t = 0; t < 8; ++t) {
        int jcol = j0 + t * 16 + n;
        float qn = sqq[bh * L_ + jcol];
#pragma unroll
        for (int r = 0; r < 8; ++r) {
            int irow = i0 + r + mofs;
            float c = acc[t][r];
            float d2 = sqk[bh * L_ + irow] + qn - 2.0f * c;
            float s = -sqrtf(fmaxf(d2, 0.0f));
            scores[((size_t)(bh * L_ + irow)) * L_ + jcol] = s;
            lsum += s;
            lsum2 += s * s;
        }
    }

    // Block reduction of score stats, one atomic pair per block.
    const int h = bh & (H_ - 1);
    red[threadIdx.x] = lsum;
    __syncthreads();
    for (int s = 128; s > 0; s >>= 1) {
        if ((int)threadIdx.x < s) red[threadIdx.x] += red[threadIdx.x + s];
        __syncthreads();
    }
    if (threadIdx.x == 0) atomicAdd(&sstats[2 * h], red[0]);
    __syncthreads();
    red[threadIdx.x] = lsum2;
    __syncthreads();
    for (int s = 128; s > 0; s >>= 1) {
        if ((int)threadIdx.x < s) red[threadIdx.x] += red[threadIdx.x + s];
        __syncthreads();
    }
    if (threadIdx.x == 0) atomicAdd(&sstats[2 * h + 1], red[0]);
}

// ------------------------------------------------------------------
// Per-row: BN(affine) + LeakyReLU(9) + mask + softmax, in place on d_out.
__global__ __launch_bounds__(128) void softmax_k(float* __restrict__ w,
                                                 const unsigned char* __restrict__ mask,
                                                 const float* __restrict__ sstats,
                                                 const float* __restrict__ gamma,
                                                 const float* __restrict__ beta) {
    __shared__ float sd[128];
    const int row = blockIdx.x;            // (b*H + h)*L + i
    const int bh = row >> 10;
    const int h = bh & (H_ - 1);
    const int b = bh >> 3;
    const int i = row & (L_ - 1);

    const float Ns = (float)((size_t)B_ * L_ * L_);
    float mean = sstats[2 * h] / Ns;
    float var = sstats[2 * h + 1] / Ns - mean * mean;
    float inv = rsqrtf(var + 1e-5f);
    float g = gamma[h], be = beta[h];

    float* rowp = w + (size_t)row * L_;
    const unsigned char* mrow = mask + ((size_t)b * L_ + i) * L_;

    float vals[8];
    float lmax = -3.402823466e38f;
#pragma unroll
    for (int e = 0; e < 8; ++e) {
        int j = e * 128 + threadIdx.x;
        float s = rowp[j];
        s = (s - mean) * inv * g + be;
        s = (s > 0.0f) ? s : 9.0f * s;
        if (!mrow[j]) s = 0.0f;
        vals[e] = s;
        lmax = fmaxf(lmax, s);
    }
    sd[threadIdx.x] = lmax;
    __syncthreads();
    for (int s = 64; s > 0; s >>= 1) {
        if ((int)threadIdx.x < s) sd[threadIdx.x] = fmaxf(sd[threadIdx.x], sd[threadIdx.x + s]);
        __syncthreads();
    }
    float gmax = sd[0];
    __syncthreads();

    float lsum = 0.0f;
#pragma unroll
    for (int e = 0; e < 8; ++e) {
        vals[e] = __expf(vals[e] - gmax);
        lsum += vals[e];
    }
    sd[threadIdx.x] = lsum;
    __syncthreads();
    for (int s = 64; s > 0; s >>= 1) {
        if ((int)threadIdx.x < s) sd[threadIdx.x] += sd[threadIdx.x + s];
        __syncthreads();
    }
    float gsum = sd[0];
    float invs = (gsum > 0.0f && gsum < 3.0e38f) ? 1.0f / gsum : 0.0f;  // NaN guard
#pragma unroll
    for (int e = 0; e < 8; ++e) {
        int j = e * 128 + threadIdx.x;
        rowp[j] = vals[e] * invs;
    }
}

// ------------------------------------------------------------------
extern "C" void kernel_launch(void* const* d_in, const int* in_sizes, int n_in,
                              void* d_out, int out_size, void* d_ws, size_t ws_size,
                              hipStream_t stream) {
    const float* Q = (const float*)d_in[0];
    const float* K = (const float*)d_in[1];
    const float* V = (const float*)d_in[2];
    const unsigned char* mask = (const unsigned char*)d_in[3];
    const int* bx = (const int*)d_in[4];
    const float* gamma = (const float*)d_in[5];
    const float* beta = (const float*)d_in[6];

    if (ws_size < WS_NEED) return;  // ~8.7 MB scratch required

    char* ws = (char*)d_ws;
    bf16_t* Kb = (bf16_t*)(ws + WS_KB16);
    bf16_t* Qb = (bf16_t*)(ws + WS_QB16);
    float* sqk = (float*)(ws + WS_SQK);
    float* sqq = (float*)(ws + WS_SQQ);
    float* stats = (float*)(ws + WS_STATS);   // [0..15] V stats, [16..31] score stats

    float* W = (float*)d_out;
    int* bxo = (int*)d_out + W_ELEMS;
    float* Vn = (float*)d_out + W_ELEMS + BX_ELEMS;

    init_stats_k<<<1, 32, 0, stream>>>(stats);

    pack_bf16_k<<<2 * (int)NROW, 64, 0, stream>>>(Q, K, Qb, Kb, sqq, sqk);

    v_stats_k<<<2048, 256, 0, stream>>>(V, stats);

    v_norm_k<<<(int)((size_t)B_ * L_ * D_ / 256), 256, 0, stream>>>(V, stats, Vn);

    bx_copy_k<<<((int)BX_ELEMS + 255) / 256, 256, 0, stream>>>(bx, bxo);

    dim3 gg(L_ / 128, L_ / 128, B_ * H_);   // (8, 8, 32)
    gemm_scores_k<<<gg, 256, 0, stream>>>(Kb, Qb, sqk, sqq, W, stats + 16);

    softmax_k<<<(int)NROW, 128, 0, stream>>>(W, mask, stats + 16, gamma, beta);
}
